// GlobalClassification_40054865002713
// MI455X (gfx1250) — matrix operand order repaired and checked by reading, not compiled, not run on hardware
//
#include <hip/hip_runtime.h>

typedef __attribute__((ext_vector_type(16))) __bf16 v16bf;
typedef __attribute__((ext_vector_type(8)))  float  v8f;

constexpr int kB = 4096;
constexpr int kD = 1024;
constexpr int kT = 768;
constexpr int kC = 16;
constexpr float kCurv     = 0.05f;
constexpr float kSc       = 0.22360679774997896f;   // sqrt(0.05)
constexpr float kEpsF     = 1e-8f;
constexpr float kAsinhMax = 11.090354888959125f;    // asinh(2^15)

// ---------------- helpers ----------------
__device__ __forceinline__ unsigned short f2bf(float f) {
  unsigned int u = __float_as_uint(f);
  u += 0x7FFFu + ((u >> 16) & 1u);      // round-to-nearest-even
  return (unsigned short)(u >> 16);
}
__device__ __forceinline__ float bf2f(unsigned short h) {
  return __uint_as_float(((unsigned int)h) << 16);
}
__device__ __forceinline__ float clampf(float x, float lo, float hi) {
  return fminf(fmaxf(x, lo), hi);
}
// 256-thread (8-wave, wave32) block reduction
__device__ __forceinline__ float blockReduceSum(float v, float* sred) {
  int lane = threadIdx.x & 31, w = threadIdx.x >> 5;
  #pragma unroll
  for (int o = 16; o > 0; o >>= 1) v += __shfl_down(v, o, 32);
  if (lane == 0) sred[w] = v;
  __syncthreads();
  int nw = blockDim.x >> 5;
  float r = (threadIdx.x < nw) ? sred[threadIdx.x] : 0.0f;
  if (w == 0) {
    #pragma unroll
    for (int o = 16; o > 0; o >>= 1) r += __shfl_down(r, o, 32);
    if (lane == 0) sred[0] = r;
  }
  __syncthreads();
  float out = sred[0];
  __syncthreads();
  return out;
}

// ---------------- fp32 -> bf16 converter ----------------
__global__ void k_cvt_bf16(const float* __restrict__ src,
                           unsigned short* __restrict__ dst, long long n) {
  long long i = (long long)blockIdx.x * blockDim.x + threadIdx.x;
  long long stride = (long long)gridDim.x * blockDim.x;
  for (; i < n; i += stride) dst[i] = f2bf(src[i]);
}

// ---------------- BF16 WMMA GEMM:  C[M,N] = A[M,K] * W[N,K]^T + bias ----------------
// Block tile 128x64x32; 8 waves, each 2x2 of 16x16x32 WMMA.
// Double-buffered LDS ping-pong: one barrier per K-step; next-tile global loads
// are issued into registers before the WMMAs so their wait overlaps matrix work.
__global__ __launch_bounds__(256) void k_gemm_bf16(
    const unsigned short* __restrict__ A,   // M x K row-major bf16
    const unsigned short* __restrict__ W,   // N x K row-major bf16
    const float* __restrict__ bias,         // N
    float* __restrict__ Cf,                 // fp32 out (or null)
    unsigned short* __restrict__ Ch,        // bf16 out (or null)
    int M, int N, int K) {
  __shared__ unsigned short sA[2][128][40];    // pad rows to 80B stride
  __shared__ unsigned short sB[2][64][40];
  const int bm = blockIdx.x * 128;
  const int bn = blockIdx.y * 64;
  const int tid = threadIdx.x;
  const int lane = tid & 31;
  const int wave = tid >> 5;
  const int wm = (wave & 3) * 32;           // 4 waves along M
  const int wn = (wave >> 2) * 32;          // 2 waves along N
  const int khalf = lane >> 4;
  const int l16 = lane & 15;

  // staging addresses: each thread owns 32B of A (two chunks) and 16B of B
  const int arow = tid >> 1;                // 0..127
  const int acol = (tid & 1) * 16;          // 0 or 16
  const int brow = tid >> 2;                // 0..63
  const int bcol = (tid & 3) * 8;           // 0,8,16,24
  const unsigned short* pA = A + (size_t)(bm + arow) * K + acol;
  const unsigned short* pB = W + (size_t)(bn + brow) * K + bcol;

  v8f acc[2][2] = {};
  const int steps = K >> 5;

  // ---- prologue: stage K-slab 0 into buffer 0 ----
  uint4 ra0 = *(const uint4*)pA;
  uint4 ra1 = *(const uint4*)(pA + 8);
  uint4 rb  = *(const uint4*)pB;
  pA += 32; pB += 32;
  *(uint4*)&sA[0][arow][acol]     = ra0;
  *(uint4*)&sA[0][arow][acol + 8] = ra1;
  *(uint4*)&sB[0][brow][bcol]     = rb;
  __syncthreads();

  for (int kk = 0; kk < steps; ++kk) {
    const int cur = kk & 1;
    const bool hasNext = (kk + 1) < steps;
    if (hasNext) {                          // issue next-slab loads early
      ra0 = *(const uint4*)pA;
      ra1 = *(const uint4*)(pA + 8);
      rb  = *(const uint4*)pB;
      pA += 32; pB += 32;
    }

    // ---- fragments per CDNA5 WMMA VGPR layouts ----
    v16bf afrag[2], bfrag[2];
    #pragma unroll
    for (int tm = 0; tm < 2; ++tm) {
      const unsigned short* p = &sA[cur][wm + tm * 16 + l16][khalf * 8];
      __builtin_memcpy(&afrag[tm], p, 16);                                    // K = khalf*8 .. +7
      __builtin_memcpy(reinterpret_cast<char*>(&afrag[tm]) + 16, p + 16, 16); // K = 16+khalf*8 ..
    }
    #pragma unroll
    for (int tn = 0; tn < 2; ++tn) {
      const unsigned short* p = &sB[cur][wn + tn * 16 + l16][khalf * 8];
      __builtin_memcpy(&bfrag[tn], p, 16);                                    // K = khalf*8 .. +7
      __builtin_memcpy(reinterpret_cast<char*>(&bfrag[tn]) + 16, p + 16, 16); // K = 16+khalf*8 ..
    }
    #pragma unroll
    for (int tm = 0; tm < 2; ++tm)
      #pragma unroll
      for (int tn = 0; tn < 2; ++tn)
        acc[tm][tn] = __builtin_amdgcn_wmma_f32_16x16x32_bf16(
            false, afrag[tm], false, bfrag[tn], (short)0, acc[tm][tn], false, false);

    if (hasNext) {                          // wait-for-loads overlaps the WMMAs above
      const int nxt = cur ^ 1;
      *(uint4*)&sA[nxt][arow][acol]     = ra0;
      *(uint4*)&sA[nxt][arow][acol + 8] = ra1;
      *(uint4*)&sB[nxt][brow][bcol]     = rb;
    }
    __syncthreads();  // single barrier: guards reads-of-cur vs next writes AND writes-of-nxt vs next reads
  }

  // ---- epilogue: C/D layout M = r + 8*khalf, N = l16 ----
  #pragma unroll
  for (int tm = 0; tm < 2; ++tm) {
    #pragma unroll
    for (int tn = 0; tn < 2; ++tn) {
      #pragma unroll
      for (int r = 0; r < 8; ++r) {
        int m = bm + wm + tm * 16 + r + 8 * khalf;
        int n = bn + wn + tn * 16 + l16;
        float v = acc[tm][tn][r] + bias[n];
        if (Cf) Cf[(size_t)m * N + n] = v;
        if (Ch) Ch[(size_t)m * N + n] = f2bf(v);
      }
    }
  }
}

// ---------------- p_logmap0 rows + stack into X (fp32 + bf16) ----------------
__global__ __launch_bounds__(256) void k_pre(
    const float* __restrict__ ohyp, const float* __restrict__ fhyp,
    const float* __restrict__ oeuc, const float* __restrict__ feuc,
    float* __restrict__ Xf, unsigned short* __restrict__ Xh) {
  __shared__ float sred[32];
  int b = blockIdx.x, tid = threadIdx.x;
  const float* srcs[4] = {ohyp, fhyp, oeuc, feuc};
  for (int s = 0; s < 4; ++s) {
    const float* src = srcs[s] + (size_t)b * kD;
    float v[4]; float p = 0.f;
    #pragma unroll
    for (int i = 0; i < 4; ++i) { v[i] = src[tid + i * 256]; p += v[i] * v[i]; }
    float scale = 1.0f;
    if (s < 2) {
      float n2 = blockReduceSum(p, sred);
      float n = fmaxf(sqrtf(n2), 1e-15f);
      float x = clampf(kSc * n, -1.0f + 1e-5f, 1.0f - 1e-5f);
      float at = 0.5f * (log1pf(x) - log1pf(-x));     // artanh
      scale = at / (n * kSc);
    }
    size_t row = ((size_t)b * 4 + s) * kD;
    #pragma unroll
    for (int i = 0; i < 4; ++i) {
      float y = v[i] * scale;
      Xf[row + tid + i * 256] = y;
      Xh[row + tid + i * 256] = f2bf(y);
    }
  }
}

// ---------------- attention core (S=4, hd=256): one block per (b,h) ----------------
__global__ __launch_bounds__(128) void k_attn(const unsigned short* __restrict__ QKV,
                                              unsigned short* __restrict__ O) {
  __shared__ float sq[4][256], sk[4][256], sv[4][256];
  __shared__ float sdot[16][8];
  __shared__ float sa[4][4];
  int bh = blockIdx.x, tid = threadIdx.x;
  int b = bh >> 2, h = bh & 3;
  for (int e = tid; e < 1024; e += 128) {
    int i = e >> 8, d = e & 255;
    size_t base = ((size_t)(b * 4 + i)) * (3 * kD) + h * 256 + d;
    sq[i][d] = bf2f(QKV[base]);
    sk[i][d] = bf2f(QKV[base + kD]);
    sv[i][d] = bf2f(QKV[base + 2 * kD]);
  }
  __syncthreads();
  { // 16 score pairs, 8 lanes each
    int p = tid >> 3, sub = tid & 7;
    int i = p >> 2, j = p & 3;
    float part = 0.f;
    for (int d = sub; d < 256; d += 8) part += sq[i][d] * sk[j][d];
    sdot[p][sub] = part;
  }
  __syncthreads();
  if (tid < 16) {
    float s = 0.f;
    #pragma unroll
    for (int u = 0; u < 8; ++u) s += sdot[tid][u];
    sa[tid >> 2][tid & 3] = s * (1.0f / 16.0f);       // / sqrt(256)
  }
  __syncthreads();
  if (tid < 4) {
    float m = sa[tid][0];
    #pragma unroll
    for (int j = 1; j < 4; ++j) m = fmaxf(m, sa[tid][j]);
    float e[4], sum = 0.f;
    #pragma unroll
    for (int j = 0; j < 4; ++j) { e[j] = expf(sa[tid][j] - m); sum += e[j]; }
    #pragma unroll
    for (int j = 0; j < 4; ++j) sa[tid][j] = e[j] / sum;
  }
  __syncthreads();
  for (int e = tid; e < 1024; e += 128) {
    int i = e >> 8, d = e & 255;
    float o = sa[i][0] * sv[0][d] + sa[i][1] * sv[1][d] +
              sa[i][2] * sv[2][d] + sa[i][3] * sv[3][d];
    O[((size_t)(b * 4 + i)) * kD + h * 256 + d] = f2bf(o);
  }
}

// ---------------- residual + LayerNorm (in-place into gf); emit oeu/feu ----------------
__global__ __launch_bounds__(256) void k_ln(float* __restrict__ Xf, const float* __restrict__ P,
                                            const float* __restrict__ w, const float* __restrict__ bvec,
                                            float* __restrict__ out_oeu, float* __restrict__ out_feu) {
  __shared__ float sred[32];
  int r = blockIdx.x, tid = threadIdx.x;
  size_t base = (size_t)r * kD;
  float v[4]; float s1 = 0.f;
  #pragma unroll
  for (int i = 0; i < 4; ++i) { int d = tid + i * 256; v[i] = Xf[base + d] + P[base + d]; s1 += v[i]; }
  float mu = blockReduceSum(s1, sred) * (1.0f / kD);
  float s2 = 0.f;
  #pragma unroll
  for (int i = 0; i < 4; ++i) { float t = v[i] - mu; s2 += t * t; }
  float var = blockReduceSum(s2, sred) * (1.0f / kD);
  float rstd = rsqrtf(var + 1e-5f);
  int s = r & 3, b = r >> 2;
  #pragma unroll
  for (int i = 0; i < 4; ++i) {
    int d = tid + i * 256;
    float y = (v[i] - mu) * rstd * w[d] + bvec[d];
    Xf[base + d] = y;
    if (s == 2) out_oeu[(size_t)b * kD + d] = y;
    if (s == 3) out_feu[(size_t)b * kD + d] = y;
  }
}

// ---------------- oh/fh = l_expmap0 rows; hyp_pt = p_project(p_expmap0(l_logmap0(oh))) ----------------
__global__ __launch_bounds__(256) void k_post(const float* __restrict__ gf,
                                              float* __restrict__ out_oh, float* __restrict__ out_fh,
                                              float* __restrict__ hyp_pt) {
  __shared__ float sred[32];
  int b = blockIdx.x, tid = threadIdx.x;
  { // row s=0 -> oh and hyp_pt (all scalar multiples of gf row)
    size_t base = ((size_t)b * 4) * kD;
    float v[4]; float p = 0.f;
    #pragma unroll
    for (int i = 0; i < 4; ++i) { v[i] = gf[base + tid + i * 256]; p += v[i] * v[i]; }
    float n = sqrtf(blockReduceSum(p, sred));
    float rc = kSc * n;
    float si = clampf(rc, kEpsF, kAsinhMax);
    float alpha = sinhf(si) / fmaxf(rc, kEpsF);             // l_expmap0 scale
    float rc2 = kSc * alpha * n;                            // l_logmap0 of oh
    float beta = (asinhf(rc2) / fmaxf(rc2, kEpsF)) * alpha;
    float nu = fmaxf(beta * n, 1e-15f);                     // p_expmap0
    float gamma = (tanhf(kSc * nu) / (kSc * nu)) * beta;
    float npt = fmaxf(gamma * n, 1e-15f);                   // p_project
    float maxn = 0.996f / kSc;
    if (npt > maxn) gamma *= maxn / npt;
    #pragma unroll
    for (int i = 0; i < 4; ++i) {
      int d = tid + i * 256;
      out_oh[(size_t)b * kD + d] = alpha * v[i];
      hyp_pt[(size_t)b * kD + d] = gamma * v[i];
    }
  }
  { // row s=1 -> fh
    size_t base = ((size_t)b * 4 + 1) * kD;
    float v[4]; float p = 0.f;
    #pragma unroll
    for (int i = 0; i < 4; ++i) { v[i] = gf[base + tid + i * 256]; p += v[i] * v[i]; }
    float n = sqrtf(blockReduceSum(p, sred));
    float rc = kSc * n;
    float si = clampf(rc, kEpsF, kAsinhMax);
    float alpha = sinhf(si) / fmaxf(rc, kEpsF);
    #pragma unroll
    for (int i = 0; i < 4; ++i)
      out_fh[(size_t)b * kD + tid + i * 256] = alpha * v[i];
  }
}

// ---------------- entailment penalty per row (oxy_angle - half_aperture) ----------------
__device__ __forceinline__ float penalty(float sumU2, float dotUY, float sumY2) {
  float nu = sqrtf(sumU2);
  float rc = kSc * nu;
  float si = clampf(rc, kEpsF, kAsinhMax);
  float se = sinhf(si) / fmaxf(rc, kEpsF);      // l_expmap0 scale for x = se*u
  float sumx2 = se * se * sumU2;
  float normx = se * nu;
  float dotxy = se * dotUY;
  float xt = sqrtf(1.0f / kCurv + sumx2);
  float yt = sqrtf(1.0f / kCurv + sumY2);
  float cxy = kCurv * (dotxy - xt * yt);
  float num = yt + cxy * xt;
  float den = sqrtf(fmaxf(cxy * cxy - 1.0f, kEpsF));
  float ain = clampf(num / (normx * den + kEpsF), -1.0f + kEpsF, 1.0f - kEpsF);
  float ang = acosf(ain);
  float asi = clampf(0.2f / (normx * kSc + kEpsF), -1.0f + kEpsF, 1.0f - kEpsF);
  float ha = asinf(asi);
  return fmaxf(ang - ha, 0.0f);
}

__global__ __launch_bounds__(256) void k_loss(const float* __restrict__ OF, const float* __restrict__ FF,
                                              const float* __restrict__ oh, const float* __restrict__ fh,
                                              const int* __restrict__ mask,
                                              float* __restrict__ penO, float* __restrict__ penF,
                                              float* __restrict__ msum) {
  __shared__ float sred[32];
  int b = blockIdx.x, tid = threadIdx.x;
  size_t base = (size_t)b * kD;
  float pu = 0, puf = 0, pf = 0, pv = 0, pvo = 0, po = 0;
  #pragma unroll
  for (int i = 0; i < 4; ++i) {
    int d = tid + i * 256;
    float u = OF[base + d], f = fh[base + d], w = FF[base + d], o = oh[base + d];
    pu += u * u; puf += u * f; pf += f * f;
    pv += w * w; pvo += w * o; po += o * o;
  }
  float su2 = blockReduceSum(pu, sred);
  float duf = blockReduceSum(puf, sred);
  float sf2 = blockReduceSum(pf, sred);
  float sv2 = blockReduceSum(pv, sred);
  float dvo = blockReduceSum(pvo, sred);
  float so2 = blockReduceSum(po, sred);
  if (tid == 0) {
    penO[b] = penalty(su2, duf, sf2);
    float m = (float)mask[b];
    penF[b] = penalty(sv2, dvo, so2) * m;
    msum[b] = m;
  }
}

// ---------------- MLR precompute per class ----------------
__global__ __launch_bounds__(256) void k_mlrprep(const float* __restrict__ mlr_a,
                                                 const float* __restrict__ mlr_p,
                                                 float* __restrict__ Pp, float* __restrict__ Ap,
                                                 float* __restrict__ Scal) {
  __shared__ float sred[32];
  int c = blockIdx.x, tid = threadIdx.x;
  size_t base = (size_t)c * kD;
  float pv[4], av[4]; float pp2 = 0.f;
  #pragma unroll
  for (int i = 0; i < 4; ++i) {
    int d = tid + i * 256;
    pv[i] = mlr_p[base + d]; av[i] = mlr_a[base + d];
    pp2 += pv[i] * pv[i];
  }
  float sp2 = blockReduceSum(pp2, sred);
  float np = fmaxf(sqrtf(sp2), 1e-15f);
  float sclp = tanhf(kSc * np) / (kSc * np);        // p_expmap0 scale
  float x2 = sclp * sclp * sp2;                     // sum(p_p^2)
  float conf = 1.0f - kCurv * x2;
  float ppa = 0.f, pa2 = 0.f;
  #pragma unroll
  for (int i = 0; i < 4; ++i) {
    int d = tid + i * 256;
    float ppv = sclp * pv[i], apv = conf * av[i];
    Pp[base + d] = ppv; Ap[base + d] = apv;
    ppa += ppv * apv; pa2 += apv * apv;
  }
  float spa = blockReduceSum(ppa, sred);
  float sa2 = blockReduceSum(pa2, sred);
  if (tid == 0) {
    float an = sqrtf(sa2);
    float lam = 2.0f / (1.0f - kCurv * x2);
    Scal[c * 8 + 0] = x2;
    Scal[c * 8 + 1] = conf;            // beta = 1 - c*||-p_p||^2
    Scal[c * 8 + 2] = -spa;            // dot(-p_p, a_p)
    Scal[c * 8 + 3] = an;              // ||a_p||
    Scal[c * 8 + 4] = lam * an / kSc;  // kk
  }
}

// ---------------- hyperbolic MLR logits (analytic Mobius reduction) ----------------
__global__ __launch_bounds__(256) void k_logits(const float* __restrict__ hyp_pt,
                                                const float* __restrict__ Pp, const float* __restrict__ Ap,
                                                const float* __restrict__ Scal,
                                                float* __restrict__ out_logits) {
  __shared__ float sred[32];
  int b = blockIdx.x, tid = threadIdx.x;
  size_t base = (size_t)b * kD;
  float y[4]; float py2 = 0.f;
  #pragma unroll
  for (int i = 0; i < 4; ++i) { y[i] = hyp_pt[base + tid + i * 256]; py2 += y[i] * y[i]; }
  float y2 = blockReduceSum(py2, sred);
  for (int c = 0; c < kC; ++c) {
    size_t cb = (size_t)c * kD;
    float pxy = 0.f, pya = 0.f;
    #pragma unroll
    for (int i = 0; i < 4; ++i) {
      int d = tid + i * 256;
      pxy += Pp[cb + d] * y[i];
      pya += Ap[cb + d] * y[i];
    }
    float xy = -blockReduceSum(pxy, sred);   // dot(-p_p, y)
    float ya = blockReduceSum(pya, sred);    // dot(a_p, y)
    if (tid == 0) {
      float x2 = Scal[c * 8 + 0], beta = Scal[c * 8 + 1], dpa = Scal[c * 8 + 2];
      float an = Scal[c * 8 + 3], kk = Scal[c * 8 + 4];
      float alpha = 1.0f + 2.0f * kCurv * xy + kCurv * y2;
      float den = 1.0f + 2.0f * kCurv * xy + kCurv * kCurv * x2 * y2 + 1e-5f;
      float dma = (alpha * dpa + beta * ya) / den;          // dot(mob, a_p)
      float num2 = 2.0f * kSc * dma;
      float sm2 = (alpha * alpha * x2 + 2.0f * alpha * beta * xy + beta * beta * y2) / (den * den);
      float den2 = an * (1.0f - kCurv * sm2);
      out_logits[(size_t)b * kC + c] = kk * asinhf(num2 / den2);
    }
  }
}

// ---------------- final deterministic loss reduction ----------------
__global__ __launch_bounds__(256) void k_final(const float* __restrict__ penO,
                                               const float* __restrict__ penF,
                                               const float* __restrict__ msum,
                                               float* __restrict__ out_loss) {
  __shared__ float sred[32];
  int tid = threadIdx.x;
  float po = 0.f, pf = 0.f, pm = 0.f;
  for (int i = tid; i < kB; i += 256) { po += penO[i]; pf += penF[i]; pm += msum[i]; }
  po = blockReduceSum(po, sred);
  pf = blockReduceSum(pf, sred);
  pm = blockReduceSum(pm, sred);
  if (tid == 0) {
    float loss = po / (float)kB;
    if (pm > 0.0f) loss += pf / fmaxf(pm, 1.0f);
    out_loss[0] = loss;
  }
}

// ---------------- host orchestration ----------------
extern "C" void kernel_launch(void* const* d_in, const int* in_sizes, int n_in,
                              void* d_out, int out_size, void* d_ws, size_t ws_size,
                              hipStream_t stream) {
  (void)in_sizes; (void)n_in; (void)out_size;
  const float* order_hyp  = (const float*)d_in[0];
  const float* family_hyp = (const float*)d_in[1];
  const float* order_euc  = (const float*)d_in[2];
  const float* family_euc = (const float*)d_in[3];
  const float* order      = (const float*)d_in[4];
  const float* family     = (const float*)d_in[5];
  const int*   mask       = (const int*)  d_in[6];
  const float* in_proj_w  = (const float*)d_in[7];
  const float* in_proj_b  = (const float*)d_in[8];
  const float* out_proj_w = (const float*)d_in[9];
  const float* out_proj_b = (const float*)d_in[10];
  const float* ln_w       = (const float*)d_in[11];
  const float* ln_b       = (const float*)d_in[12];
  const float* to_w       = (const float*)d_in[13];
  const float* to_b       = (const float*)d_in[14];
  const float* tf_w       = (const float*)d_in[15];
  const float* tf_b       = (const float*)d_in[16];
  const float* mlr_a      = (const float*)d_in[17];
  const float* mlr_p      = (const float*)d_in[18];

  float* out = (float*)d_out;
  float* out_logits = out;
  float* out_loss   = out + (size_t)kB * kC;
  float* out_oh     = out_loss + 1;
  float* out_fh     = out_oh + (size_t)kB * kD;
  float* out_oeu    = out_fh + (size_t)kB * kD;
  float* out_feu    = out_oeu + (size_t)kB * kD;

  size_t off = 0;
  auto take = [&](size_t elems, size_t esz) -> void* {
    void* p = (char*)d_ws + off;
    off += ((elems * esz + 255) & ~(size_t)255);
    return p;
  };
  float*          Xf   = (float*)         take((size_t)kB * 4 * kD, 4);
  unsigned short* Xh   = (unsigned short*)take((size_t)kB * 4 * kD, 2);
  unsigned short* Wqkv = (unsigned short*)take((size_t)3 * kD * kD, 2);
  unsigned short* Wout = (unsigned short*)take((size_t)kD * kD, 2);
  unsigned short* Wto  = (unsigned short*)take((size_t)kD * kT, 2);
  unsigned short* Wtf  = (unsigned short*)take((size_t)kD * kT, 2);
  unsigned short* OrdH = (unsigned short*)take((size_t)kB * kT, 2);
  unsigned short* FamH = (unsigned short*)take((size_t)kB * kT, 2);
  unsigned short* QKVh = (unsigned short*)take((size_t)kB * 4 * 3 * kD, 2);
  unsigned short* Oh   = (unsigned short*)take((size_t)kB * 4 * kD, 2);
  float*          P    = (float*)         take((size_t)kB * 4 * kD, 4);
  float*          OF   = (float*)         take((size_t)kB * kD, 4);
  float*          FF   = (float*)         take((size_t)kB * kD, 4);
  float*          Hyp  = (float*)         take((size_t)kB * kD, 4);
  float*          Pp   = (float*)         take((size_t)kC * kD, 4);
  float*          Ap   = (float*)         take((size_t)kC * kD, 4);
  float*          Scal = (float*)         take((size_t)kC * 8, 4);
  float*          PenO = (float*)         take((size_t)kB, 4);
  float*          PenF = (float*)         take((size_t)kB, 4);
  float*          Msum = (float*)         take((size_t)kB, 4);
  if (off > ws_size) return;  // insufficient scratch

  // weight / token bf16 conversions
  k_cvt_bf16<<<2048, 256, 0, stream>>>(in_proj_w,  Wqkv, (long long)3 * kD * kD);
  k_cvt_bf16<<<2048, 256, 0, stream>>>(out_proj_w, Wout, (long long)kD * kD);
  k_cvt_bf16<<<1024, 256, 0, stream>>>(to_w, Wto, (long long)kD * kT);
  k_cvt_bf16<<<1024, 256, 0, stream>>>(tf_w, Wtf, (long long)kD * kT);
  k_cvt_bf16<<<2048, 256, 0, stream>>>(order,  OrdH, (long long)kB * kT);
  k_cvt_bf16<<<2048, 256, 0, stream>>>(family, FamH, (long long)kB * kT);

  // p_logmap0 + stack
  k_pre<<<kB, 256, 0, stream>>>(order_hyp, family_hyp, order_euc, family_euc, Xf, Xh);

  // QKV: (16384 x 1024) @ (1024 x 3072)
  k_gemm_bf16<<<dim3((kB * 4) / 128, (3 * kD) / 64), 256, 0, stream>>>(
      Xh, Wqkv, in_proj_b, nullptr, QKVh, kB * 4, 3 * kD, kD);
  // attention core
  k_attn<<<kB * 4, 128, 0, stream>>>(QKVh, Oh);
  // out projection: (16384 x 1024) @ (1024 x 1024)
  k_gemm_bf16<<<dim3((kB * 4) / 128, kD / 64), 256, 0, stream>>>(
      Oh, Wout, out_proj_b, P, nullptr, kB * 4, kD, kD);
  // residual + layernorm (in-place into Xf); emits oeu/feu
  k_ln<<<kB * 4, 256, 0, stream>>>(Xf, P, ln_w, ln_b, out_oeu, out_feu);
  // oh/fh/hyp_pt
  k_post<<<kB, 256, 0, stream>>>(Xf, out_oh, out_fh, Hyp);

  // order/family feature GEMMs: (4096 x 768) @ (768 x 1024)
  k_gemm_bf16<<<dim3(kB / 128, kD / 64), 256, 0, stream>>>(
      OrdH, Wto, to_b, OF, nullptr, kB, kD, kT);
  k_gemm_bf16<<<dim3(kB / 128, kD / 64), 256, 0, stream>>>(
      FamH, Wtf, tf_b, FF, nullptr, kB, kD, kT);

  // loss rows -> deterministic final reduction
  k_loss<<<kB, 256, 0, stream>>>(OF, FF, out_oh, out_fh, mask, PenO, PenF, Msum);
  k_final<<<1, 256, 0, stream>>>(PenO, PenF, Msum, out_loss);

  // hyperbolic MLR
  k_mlrprep<<<kC, 256, 0, stream>>>(mlr_a, mlr_p, Pp, Ap, Scal);
  k_logits<<<kB, 256, 0, stream>>>(Hyp, Pp, Ap, Scal, out_logits);
}
